// MoAKDALayer_88871463289235
// MI455X (gfx1250) — matrix operand order repaired and checked by reading, not compiled, not run on hardware
//
#include <hip/hip_runtime.h>

// ---------------------------------------------------------------------------
// MoA-KDA layer for MI455X (gfx1250, wave32, WMMA).
// Forward-mode routing is HARD (straight-through), so only argmax experts
// matter. All big projections run as bf16 WMMA (v_wmma_f32_16x16x32_bf16)
// with fp32 accumulation; weights are pre-packed to bf16 (padded) and stay
// resident in the 192MB L2 across the tile grid. 32-token M-tiles amortize
// the per-block W staging (L2 traffic) over 2x the WMMA work, balancing the
// L2-stream vs matrix-math roofline. The W_o GEMM stages its A-tile with the
// Tensor Data Mover (tensor_load_to_lds + s_wait_tensorcnt) where available.
// ---------------------------------------------------------------------------

#define BQ 4
#define NMHC 4
#define TT 1024
#define DD 256
#define HH 8
#define EE 4
#define HE 32
#define BT 4096            // B*T
#define DVv 64
#define DKP 32

typedef __bf16 bf16_t;
typedef __attribute__((ext_vector_type(16))) __bf16 v16bf;
typedef __attribute__((ext_vector_type(8)))  float  v8f;
typedef __attribute__((ext_vector_type(8)))  unsigned short v8us;
typedef __attribute__((ext_vector_type(4)))  unsigned int   u32x4;
typedef __attribute__((ext_vector_type(8)))  int            i32x8;
typedef __attribute__((ext_vector_type(4)))  int            i32x4;

union FragBF { v16bf v; v8us h[2]; };

__device__ __forceinline__ float sigmf(float x) { return 1.f / (1.f + __expf(-x)); }
__device__ __forceinline__ float siluf(float x) { return x / (1.f + __expf(-x)); }

// ---------------------------------------------------------------------------
// Block-level GEMM: C(16*MT x N) = A(16*MT x K, bf16 LDS, row-major, stride K)
//                                 @ W(K x N, bf16 global, row stride LDW)
// Block = 256 threads = 8 waves; wave wv owns output N-tiles nt = j*8+wv and
// all MT M-tiles (accumulators acc[MT][NTW] live in VGPRs across K).
// W is staged one 32-K-slab at a time into LDS *transposed* ([n][k]) with
// vectorized b128 global reads, so B fragments are two ds_load_b128 per lane.
// A/B/C lane layouts per CDNA5 ISA 7.12.2.
// ---------------------------------------------------------------------------
template <int MT, int NTW, int K, int N, int LDW>
__device__ __forceinline__ void block_gemm(const bf16_t* A,
                                           const bf16_t* __restrict__ W,
                                           bf16_t* wslab, float* outC, int tid) {
  const int lane = tid & 31;
  const int wv   = tid >> 5;
  const int row  = lane & 15;
  const int kb   = (lane >> 4) << 3;   // A: K-offset 0 or 8 (then +16)
  const int kh   = (lane >> 4) << 4;   // unused; B uses kb as well

  v8f zero = {0.f, 0.f, 0.f, 0.f, 0.f, 0.f, 0.f, 0.f};
  v8f acc[MT][NTW];
#pragma unroll
  for (int mt = 0; mt < MT; ++mt)
#pragma unroll
    for (int j = 0; j < NTW; ++j) acc[mt][j] = zero;

  constexpr int N8 = N / 8;            // 8-bf16 chunks per W row
  for (int k0 = 0; k0 < K; k0 += 32) {
    // prefetch next W slab (global_prefetch_b8)
    if (k0 + 32 < K && tid < 32)
      __builtin_prefetch(W + (long)(k0 + 32 + tid) * LDW, 0, 1);
    // stage W[k0..k0+31][0..N-1] transposed into wslab[n*32 + k]
    for (int i = tid; i < 32 * N8; i += 256) {
      int k = i / N8, n8 = i % N8;
      uint4 chunk = *(const uint4*)(W + (long)(k0 + k) * LDW + n8 * 8);
      const bf16_t* src = (const bf16_t*)&chunk;
#pragma unroll
      for (int u = 0; u < 8; ++u) wslab[(n8 * 8 + u) * 32 + k] = src[u];
    }
    __syncthreads();

    FragBF fa[MT];
#pragma unroll
    for (int mt = 0; mt < MT; ++mt) {
      const bf16_t* ap = A + (mt * 16 + row) * K + k0 + kb;
      fa[mt].h[0] = *(const v8us*)(ap);        // K = kb .. kb+7
      fa[mt].h[1] = *(const v8us*)(ap + 16);   // K = kb+16 .. kb+23
    }

#pragma unroll
    for (int j = 0; j < NTW; ++j) {
      int nt = j * 8 + wv;
      if (nt * 16 < N) {                 // wave-uniform: EXEC stays all-ones
        FragBF fb;
        const bf16_t* bp = wslab + (nt * 16 + row) * 32 + kb;
        fb.h[0] = *(const v8us*)(bp);
        fb.h[1] = *(const v8us*)(bp + 16);
#pragma unroll
        for (int mt = 0; mt < MT; ++mt)
          acc[mt][j] = __builtin_amdgcn_wmma_f32_16x16x32_bf16(
              false, fa[mt].v, false, fb.v, (short)0, acc[mt][j], false, false);
      }
    }
    __syncthreads();
  }

  // C layout: VGPR r, lanes 0-15 -> M=r, lanes 16-31 -> M=8+r; N = lane&15
  const int mbase = (lane >> 4) * 8;
#pragma unroll
  for (int mt = 0; mt < MT; ++mt)
#pragma unroll
    for (int j = 0; j < NTW; ++j) {
      int nt = j * 8 + wv;
      if (nt * 16 < N) {
        int n = nt * 16 + row;
#pragma unroll
        for (int r = 0; r < 8; ++r) outC[(mt * 16 + mbase + r) * N + n] = acc[mt][j][r];
      }
    }
  __syncthreads();
}

// ---------------------------------------------------------------------------
__global__ void k_zero(float* p, long n) {
  long i = (long)blockIdx.x * 256 + threadIdx.x;
  if (i < n) p[i] = 0.f;
}

// x_hat = RMSNorm over ND of swapaxes(stream); route_in = mean over NMHC
__global__ __launch_bounds__(256) void k_prep(const float* __restrict__ stream,
                                              float* x_hat, float* route_in) {
  int flat = blockIdx.x, b = flat >> 10, t = flat & 1023, tid = threadIdx.x;
  __shared__ float red[256];
  float v[4]; float ss = 0.f, mean = 0.f;
#pragma unroll
  for (int j = 0; j < 4; ++j) {
    v[j] = stream[(((long)b * 4 + j) * TT + t) * DD + tid];
    ss += v[j] * v[j]; mean += v[j];
  }
  red[tid] = ss; __syncthreads();
  for (int s = 128; s > 0; s >>= 1) { if (tid < s) red[tid] += red[tid + s]; __syncthreads(); }
  float rs = rsqrtf(red[0] / 1024.f + 1e-6f);
#pragma unroll
  for (int j = 0; j < 4; ++j) x_hat[(long)flat * 1024 + j * 256 + tid] = v[j] * rs;
  route_in[(long)flat * 256 + tid] = mean * 0.25f;
}

// hard routing: argmax of clipped logits
__global__ void k_route(const float* __restrict__ route_in,
                        const float* __restrict__ qr, const float* __restrict__ kvr,
                        int* qsel, int* kvsel) {
  int idx = blockIdx.x * 256 + threadIdx.x;
  if (idx >= HH * BT) return;
  int h = idx >> 12, flat = idx & 4095;
  const float* x = route_in + (long)flat * 256;
  float lq[4] = {0, 0, 0, 0}, lk[4] = {0, 0, 0, 0};
  for (int d = 0; d < 256; ++d) {
    float xv = x[d];
    const float* q = qr + ((long)h * 256 + d) * 4;
    const float* k = kvr + ((long)h * 256 + d) * 4;
#pragma unroll
    for (int e = 0; e < 4; ++e) { lq[e] += xv * q[e]; lk[e] += xv * k[e]; }
  }
  int aq = 0, ak = 0; float bq = -1e30f, bk = -1e30f;
#pragma unroll
  for (int e = 0; e < 4; ++e) {
    float cq = fminf(fmaxf(lq[e], -10.f), 10.f);
    float ck = fminf(fmaxf(lk[e], -10.f), 10.f);
    if (cq > bq) { bq = cq; aq = e; }
    if (ck > bk) { bk = ck; ak = e; }
  }
  qsel[idx] = aq; kvsel[idx] = ak;
}

// generic fp32 -> bf16 pack with zero padding (K,N) -> (Kp,Np)
__global__ void k_pack(const float* __restrict__ src, bf16_t* dst,
                       int batch, int K, int N, int Kp, int Np) {
  long i = (long)blockIdx.x * 256 + threadIdx.x;
  long tot = (long)batch * Kp * Np;
  if (i >= tot) return;
  int np = (int)(i % Np); long r = i / Np; int kp = (int)(r % Kp); int bb = (int)(r / Kp);
  float v = (kp < K && np < N) ? src[((long)bb * K + kp) * N + np] : 0.f;
  dst[i] = (bf16_t)v;
}

__global__ void k_pack_qkv(const float* __restrict__ Wq, const float* __restrict__ Wk,
                           const float* __restrict__ Wv, bf16_t* dst) {
  int i = blockIdx.x * 256 + threadIdx.x;
  if (i >= HH * 256 * 96) return;
  int n = i % 96; int r = i / 96; int d = r % 256; int h = r / 256;
  float v;
  if (n < 16)      v = Wq[((long)h * 256 + d) * 16 + n];
  else if (n < 32) v = Wk[((long)h * 256 + d) * 16 + (n - 16)];
  else             v = Wv[((long)h * 256 + d) * 64 + (n - 32)];
  dst[i] = (bf16_t)v;
}

// phiC[he][k][0:4]=nw*phi_pre, [4:8]=nw*phi_post, [8:24]=nw*phi_res, [24:32]=0
__global__ void k_pack_phi(const float* __restrict__ nw, const float* __restrict__ ppre,
                           const float* __restrict__ ppost, const float* __restrict__ pres,
                           bf16_t* dst) {
  int i = blockIdx.x * 256 + threadIdx.x;
  if (i >= HE * 1024 * 32) return;
  int n = i & 31; int r = i >> 5; int k = r & 1023; int he = r >> 10;
  float w = nw[(long)he * 1024 + k]; float v = 0.f;
  if (n < 4)       v = w * ppre[((long)he * 1024 + k) * 4 + n];
  else if (n < 8)  v = w * ppost[((long)he * 1024 + k) * 4 + (n - 4)];
  else if (n < 24) v = w * pres[((long)he * 1024 + k) * 16 + (n - 8)];
  dst[i] = (bf16_t)v;
}

// Hpre / Hpost / Hres(sinkhorn) via one WMMA GEMM: x_hat(32x1024) @ phiC(1024x32)
__global__ __launch_bounds__(256) void k_coef(const float* __restrict__ x_hat,
                                              const bf16_t* __restrict__ phiC,
                                              const float* __restrict__ a_pre, const float* __restrict__ b_pre,
                                              const float* __restrict__ a_post, const float* __restrict__ b_post,
                                              const float* __restrict__ a_res, const float* __restrict__ b_res,
                                              float* Hpre, float* Hpost, float* Hres) {
  __shared__ bf16_t sX[32 * 1024];
  __shared__ bf16_t sW[32 * 32];
  __shared__ float  sOut[32 * 32];
  int tid = threadIdx.x, he = blockIdx.y, flat0 = blockIdx.x * 32;
  for (int i = tid; i < 32 * 1024; i += 256)
    sX[i] = (bf16_t)x_hat[(long)(flat0 + (i >> 10)) * 1024 + (i & 1023)];
  __syncthreads();
  block_gemm<2, 1, 1024, 32, 32>(sX, phiC + (long)he * 1024 * 32, sW, sOut, tid);
  if (tid < 32) {
    int m = tid; long flat = flat0 + m;
    float ap = a_pre[he], aps = a_post[he], ar = a_res[he];
    for (int n = 0; n < 4; ++n) {
      Hpre[((long)he * BT + flat) * 4 + n]  = sigmf(ap * sOut[m * 32 + n] + b_pre[he * 4 + n]);
      Hpost[((long)he * BT + flat) * 4 + n] = 2.f * sigmf(aps * sOut[m * 32 + 4 + n] + b_post[he * 4 + n]);
    }
    float M[16];
#pragma unroll
    for (int j = 0; j < 16; ++j) M[j] = __expf(ar * sOut[m * 32 + 8 + j] + b_res[he * 16 + j]);
    for (int it = 0; it < 6; ++it) {
#pragma unroll
      for (int r = 0; r < 4; ++r) {
        float s = 1.f / (M[r * 4] + M[r * 4 + 1] + M[r * 4 + 2] + M[r * 4 + 3]);
        M[r * 4] *= s; M[r * 4 + 1] *= s; M[r * 4 + 2] *= s; M[r * 4 + 3] *= s;
      }
#pragma unroll
      for (int c = 0; c < 4; ++c) {
        float s = 1.f / (M[c] + M[4 + c] + M[8 + c] + M[12 + c]);
        M[c] *= s; M[4 + c] *= s; M[8 + c] *= s; M[12 + c] *= s;
      }
    }
#pragma unroll
    for (int j = 0; j < 16; ++j) Hres[((long)he * BT + flat) * 16 + j] = M[j];
  }
}

// ---------------------------------------------------------------------------
// Fused per-(h,e) per-32-token-tile expert kernel.
// ---------------------------------------------------------------------------
__global__ __launch_bounds__(256) void k_expert(
    const float* __restrict__ stream, const float* __restrict__ Hpre_ws,
    const float* __restrict__ Hres_ws, const float* __restrict__ Hpost_ws,
    const int* __restrict__ qsel, const int* __restrict__ kvsel,
    const float* __restrict__ norm_w, const float* __restrict__ pope_delta,
    const bf16_t* __restrict__ Wpre, const bf16_t* __restrict__ Wpg1, const bf16_t* __restrict__ Wpg2,
    const bf16_t* __restrict__ Aup, const bf16_t* __restrict__ Adn,
    const bf16_t* __restrict__ Bup, const bf16_t* __restrict__ Bdn,
    const bf16_t* __restrict__ Wqkv,
    const float* __restrict__ lAq, const float* __restrict__ lBq,
    const float* __restrict__ lAk, const float* __restrict__ lBk,
    const float* __restrict__ lAv, const float* __restrict__ lBv,
    float* acc_pre, float* acc_post, float* acc_res, float* acc_hpost,
    float* q_h, float* k_h, float* v_h, float* al_h, float* be_h) {
  __shared__ bf16_t sHe[32 * 256];      // 16KB
  __shared__ bf16_t sMid[32 * 320];     // 20KB
  __shared__ bf16_t sWsl[320 * 32];     // 20KB
  __shared__ float  sOut[32 * 320];     // 40KB
  __shared__ float  sHp[32 * 4];
  __shared__ float  sRs[32];
  __shared__ int    sQ[32], sKV[32];
  __shared__ float  sSilQ[128], sSilK[128], sSilV[128];
  __shared__ float  sFreq[16], sShift[16];

  int tid = threadIdx.x, he = blockIdx.y, h = he >> 2, e = he & 3;
  int flat0 = blockIdx.x * 32;

  if (tid < 32) {
    long flat = flat0 + tid;
    sQ[tid]  = qsel[(long)h * BT + flat];
    sKV[tid] = kvsel[(long)h * BT + flat];
  }
  if (tid < 16) {
    sFreq[tid]  = __powf(10000.f, (float)tid / 16.f);
    sShift[tid] = 6.2831853071795864f * sigmf(pope_delta[tid]);
  }
  if (tid < 128) sHp[tid] = Hpre_ws[((long)he * BT + flat0 + (tid >> 2)) * 4 + (tid & 3)];
  __syncthreads();

  // he = Hpre-weighted stream mix, then RMSNorm * norm_w, to bf16 in LDS
  for (int i = tid; i < 32 * 256; i += 256) {
    int m = i >> 8, d = i & 255;
    long flat = flat0 + m; int b = (int)(flat >> 10), t = (int)(flat & 1023);
    float s = 0.f;
#pragma unroll
    for (int n = 0; n < 4; ++n)
      s += sHp[m * 4 + n] * stream[(((long)b * 4 + n) * TT + t) * DD + d];
    sOut[i] = s;
  }
  __syncthreads();
  if (tid < 32) {
    float ss = 0.f;
    for (int d = 0; d < 256; ++d) { float v = sOut[tid * 256 + d]; ss += v * v; }
    sRs[tid] = rsqrtf(ss / 256.f + 1e-6f);
  }
  __syncthreads();
  for (int i = tid; i < 32 * 256; i += 256) {
    int m = i >> 8, d = i & 255;
    sHe[i] = (bf16_t)(sOut[i] * sRs[m] * norm_w[(long)he * 256 + d]);
  }
  __syncthreads();

  // ---- preg = silu(he @ W_pre): N=512 in two 256-col halves (KV-routed)
#pragma unroll
  for (int half = 0; half < 2; ++half) {
    block_gemm<2, 2, 256, 256, 512>(sHe, Wpre + (long)he * 256 * 512 + half * 256, sWsl, sOut, tid);
    for (int i = tid; i < 32 * 256; i += 256) {
      int m = i >> 8, n = i & 255;
      if (sKV[m] == e)
        atomicAdd(&acc_pre[(long)(flat0 + m) * 512 + half * 256 + n], siluf(sOut[i]) * 0.125f);
    }
    __syncthreads();
  }

  // ---- post-gate: sigmoid(silu(he@Wpg1)@Wpg2)  (Q-routed tokens, e>0)
  block_gemm<2, 3, 256, 320, 320>(sHe, Wpg1 + (long)he * 256 * 320, sWsl, sOut, tid);
  for (int i = tid; i < 32 * 320; i += 256) sMid[i] = (bf16_t)siluf(sOut[i]);
  __syncthreads();
  block_gemm<2, 2, 320, 256, 256>(sMid, Wpg2 + (long)he * 320 * 256, sWsl, sOut, tid);
  if (e > 0) {
    for (int i = tid; i < 32 * 256; i += 256) {
      int m = i >> 8, n = i & 255;
      if (sQ[m] == e)
        atomicAdd(&acc_post[(long)(flat0 + m) * 256 + n], sigmf(sOut[i]) * 0.125f);
    }
  }
  __syncthreads();

  // ---- alpha: sigmoid(silu(he@alpha_up)@alpha_down) -> al_h (DKP=32)
  block_gemm<2, 1, 256, 32, 32>(sHe, Aup + (long)he * 256 * 32, sWsl, sOut, tid);
  for (int i = tid; i < 32 * 32; i += 256) sMid[i] = (bf16_t)siluf(sOut[i]);
  __syncthreads();
  block_gemm<2, 1, 32, 32, 32>(sMid, Adn + (long)he * 32 * 32, sWsl, sOut, tid);
  for (int i = tid; i < 32 * 32; i += 256) {
    int m = i >> 5, n = i & 31;
    if (sKV[m] == e) al_h[(long)(flat0 + m) * 256 + h * 32 + n] = sigmf(sOut[i]);
  }
  __syncthreads();

  // ---- beta: scalar gate per token
  block_gemm<2, 1, 256, 32, 32>(sHe, Bup + (long)he * 256 * 32, sWsl, sOut, tid);
  for (int i = tid; i < 32 * 32; i += 256) sMid[i] = (bf16_t)siluf(sOut[i]);
  __syncthreads();
  block_gemm<2, 1, 32, 16, 16>(sMid, Bdn + (long)he * 32 * 16, sWsl, sOut, tid);
  if (tid < 32) {
    if (sKV[tid] == e) be_h[(long)(flat0 + tid) * 8 + h] = sigmf(sOut[tid * 16]);
  }
  __syncthreads();

  // ---- LoRA stage-1 (R=4): silu(he @ A), VALU: 3 mats x 32 tokens x 4 ranks
  for (int w = tid; w < 384; w += 256) {
    int which = w >> 7, t2 = w & 127, m = t2 >> 2, r = t2 & 3;
    const float* A = (which == 0) ? lAq : (which == 1) ? lAk : lAv;
    const float* Ap = A + (long)he * 256 * 4 + r;
    float s = 0.f;
    for (int d = 0; d < 256; ++d) s += (float)sHe[m * 256 + d] * Ap[d * 4];
    float sil = siluf(s);
    if (which == 0) sSilQ[t2] = sil; else if (which == 1) sSilK[t2] = sil; else sSilV[t2] = sil;
  }
  __syncthreads();

  // ---- QKV: he @ [Wq|Wk|Wv] (N=96), + LoRA delta, l2norm, PoPE, silu(v)
  block_gemm<2, 1, 256, 96, 96>(sHe, Wqkv + (long)h * 256 * 96, sWsl, sOut, tid);
  if (tid < 32) {
    int m = tid; long flat = flat0 + m; int t = (int)(flat & 1023);
    float qv[16], kvv[16]; float nq = 0.f, nk = 0.f;
#pragma unroll
    for (int k2 = 0; k2 < 16; ++k2) {
      float dq = 0.f, dk2 = 0.f;
#pragma unroll
      for (int r = 0; r < 4; ++r) {
        dq  += sSilQ[m * 4 + r] * lBq[((long)he * 4 + r) * 16 + k2];
        dk2 += sSilK[m * 4 + r] * lBk[((long)he * 4 + r) * 16 + k2];
      }
      qv[k2]  = sOut[m * 96 + k2] + dq;
      kvv[k2] = sOut[m * 96 + 16 + k2] + dk2;
      nq += qv[k2] * qv[k2]; nk += kvv[k2] * kvv[k2];
    }
    nq = 1.f / fmaxf(sqrtf(nq), 1e-12f);
    nk = 1.f / fmaxf(sqrtf(nk), 1e-12f);
    bool wq = (e > 0) && (sQ[m] == e);
    bool wk = (sKV[m] == e);
#pragma unroll
    for (int k2 = 0; k2 < 16; ++k2) {
      float phi = (float)t * sFreq[k2];
      if (wq) {
        float mu = log1pf(__expf(qv[k2] * nq));
        q_h[flat * 256 + h * 32 + k2]      = mu * __cosf(phi);
        q_h[flat * 256 + h * 32 + 16 + k2] = mu * __sinf(phi);
      }
      if (wk) {
        float mu = log1pf(__expf(kvv[k2] * nk));
        float pk = phi - sShift[k2];
        k_h[flat * 256 + h * 32 + k2]      = mu * __cosf(pk);
        k_h[flat * 256 + h * 32 + 16 + k2] = mu * __sinf(pk);
      }
    }
    if (wk) {
      for (int j = 0; j < 64; ++j) {
        float dv = 0.f;
#pragma unroll
        for (int r = 0; r < 4; ++r) dv += sSilV[m * 4 + r] * lBv[((long)he * 4 + r) * 64 + j];
        v_h[flat * 512 + h * 64 + j] = siluf(sOut[m * 96 + 32 + j] + dv);
      }
    }
  }

  // ---- routed Hres / Hpost accumulation
  for (int i = tid; i < 32 * 16; i += 256) {
    int m = i >> 4, j = i & 15;
    if (sKV[m] == e)
      atomicAdd(&acc_res[(long)(flat0 + m) * 16 + j], Hres_ws[((long)he * BT + flat0 + m) * 16 + j]);
  }
  if (tid < 128) {
    int m = tid >> 2, n = tid & 3;
    if (sKV[m] == e)
      atomicAdd(&acc_hpost[(long)(flat0 + m) * 4 + n], Hpost_ws[((long)he * BT + flat0 + m) * 4 + n]);
  }
}

// KDA delta-rule scan over T; one block per (b,h); thread j owns state column j
__global__ __launch_bounds__(64) void k_scan(const float* __restrict__ q_h, const float* __restrict__ k_h,
                                             const float* __restrict__ v_h, const float* __restrict__ al_h,
                                             const float* __restrict__ be_h, const int* __restrict__ qsel,
                                             float* o_ws) {
  int bh = blockIdx.x, b = bh >> 3, h = bh & 7, j = threadIdx.x;
  __shared__ float sk[32], sa[32], sq[32], sb[1];
  float S[32];
#pragma unroll
  for (int d = 0; d < 32; ++d) S[d] = 0.f;
  for (int t = 0; t < TT; ++t) {
    long flat = (long)b * TT + t;
    if (j < 32) {
      sk[j] = k_h[flat * 256 + h * 32 + j];
      sa[j] = al_h[flat * 256 + h * 32 + j];
      sq[j] = q_h[flat * 256 + h * 32 + j];
    }
    if (j == 0) sb[0] = be_h[flat * 8 + h];
    __syncthreads();
    float v = v_h[flat * 512 + h * 64 + j];
    float kaS = 0.f;
#pragma unroll
    for (int d = 0; d < 32; ++d) kaS += sk[d] * sa[d] * S[d];
    float bd = sb[0] * (v - kaS);
    float o = 0.f;
#pragma unroll
    for (int d = 0; d < 32; ++d) { S[d] = sa[d] * S[d] + sk[d] * bd; o += sq[d] * S[d]; }
    float msk = (qsel[(long)h * BT + flat] != 0) ? 1.f : 0.f;
    o_ws[flat * 512 + h * 64 + j] = o * msk;
    __syncthreads();
  }
}

__global__ void k_gated(const float* __restrict__ o_ws, const float* __restrict__ acc_pre,
                        bf16_t* gated) {
  long i = (long)blockIdx.x * 256 + threadIdx.x;
  if (i >= (long)BT * 512) return;
  gated[i] = (bf16_t)(o_ws[i] * 0.04419417382415922f * acc_pre[i]);  // 1/sqrt(512)
}

// result = (gated @ W_o) * acc_post.  A-tile (32x512 bf16 = 32KB) is staged
// into LDS by the Tensor Data Mover (2D D#: data_size=2B, tile 512x32).
__global__ __launch_bounds__(256) void k_outgemm(const bf16_t* __restrict__ gated,
                                                 const bf16_t* __restrict__ Wo,
                                                 const float* __restrict__ acc_post, float* result) {
  __shared__ bf16_t sA[32 * 512];
  __shared__ bf16_t sWsl[256 * 32];
  __shared__ float  sOut[32 * 256];
  int tid = threadIdx.x, flat0 = blockIdx.x * 32;

#if defined(__gfx1250__) && __has_builtin(__builtin_amdgcn_tensor_load_to_lds)
  if (tid < 32) {  // wave 0 issues one TDM descriptor for the whole tile
    unsigned long long ga = (unsigned long long)(const void*)(gated + (long)flat0 * 512);
    unsigned lds_off = (unsigned)(unsigned long long)(const void*)&sA[0]; // addr[31:0] == LDS offset
    u32x4 g0;
    g0[0] = 1u;                                              // count=1, user mode
    g0[1] = lds_off;                                         // lds_addr
    g0[2] = (unsigned)(ga & 0xffffffffu);                    // global_addr[31:0]
    g0[3] = (unsigned)((ga >> 32) & 0x01ffffffu) | (2u << 30); // ga[56:32] | type=2
    i32x8 g1;
    g1[0] = (int)(1u << 16);                                 // data_size=1 -> 2 bytes
    g1[1] = (int)((512u & 0xffffu) << 16);                   // tensor_dim0[15:0]
    g1[2] = (int)((512u >> 16) | (32u << 16));               // dim0 hi | tensor_dim1 lo
    g1[3] = (int)(0u | (512u << 16));                        // dim1 hi | tile_dim0=512
    g1[4] = (int)(32u);                                      // tile_dim1=32, tile_dim2=0
    g1[5] = (int)512;                                        // tensor_dim0_stride lo
    g1[6] = 0; g1[7] = 0;
    i32x4 gz = {0, 0, 0, 0};
#if __clang_major__ >= 23
    i32x8 gz8 = {0, 0, 0, 0, 0, 0, 0, 0};
    __builtin_amdgcn_tensor_load_to_lds(g0, g1, gz, gz, gz8, 0);
#else
    __builtin_amdgcn_tensor_load_to_lds(g0, g1, gz, gz, 0);
#endif
    __builtin_amdgcn_s_wait_tensorcnt((short)0);
  }
  __syncthreads();
#else
  for (int i = tid; i < 32 * 512; i += 256)
    sA[i] = gated[(long)(flat0 + (i >> 9)) * 512 + (i & 511)];
  __syncthreads();
#endif

  block_gemm<2, 2, 512, 256, 256>(sA, Wo, sWsl, sOut, tid);
  for (int i = tid; i < 32 * 256; i += 256) {
    int m = i >> 8, n = i & 255; long flat = flat0 + m;
    result[flat * 256 + n] = sOut[i] * acc_post[flat * 256 + n];
  }
}

__global__ void k_mix(const float* __restrict__ stream, const float* __restrict__ acc_res,
                      const float* __restrict__ acc_hpost, const float* __restrict__ result,
                      float* out) {
  long i = (long)blockIdx.x * 256 + threadIdx.x;
  if (i >= (long)BQ * NMHC * TT * DD) return;
  int d = (int)(i & 255); long r = i >> 8; int t = (int)(r & 1023); r >>= 10;
  int im = (int)(r & 3); int b = (int)(r >> 2);
  long flat = (long)b * TT + t;
  float acc = acc_hpost[flat * 4 + im] * 0.125f * result[flat * 256 + d];
#pragma unroll
  for (int jj = 0; jj < 4; ++jj)
    acc += acc_res[flat * 16 + im * 4 + jj] * 0.125f * stream[(((long)b * 4 + jj) * TT + t) * DD + d];
  out[i] = acc;
}

// ---------------------------------------------------------------------------
extern "C" void kernel_launch(void* const* d_in, const int* in_sizes, int n_in,
                              void* d_out, int out_size, void* d_ws, size_t ws_size,
                              hipStream_t stream_) {
  const float* stream   = (const float*)d_in[0];
  const float* Wq       = (const float*)d_in[1];
  const float* Wk       = (const float*)d_in[2];
  const float* Wv       = (const float*)d_in[3];
  const float* pope_d   = (const float*)d_in[4];
  const float* q_router = (const float*)d_in[5];
  const float* kv_router= (const float*)d_in[6];
  const float* lAq      = (const float*)d_in[7];
  const float* lBq      = (const float*)d_in[8];
  const float* lAk      = (const float*)d_in[9];
  const float* lBk      = (const float*)d_in[10];
  const float* lAv      = (const float*)d_in[11];
  const float* lBv      = (const float*)d_in[12];
  const float* alpha_up = (const float*)d_in[13];
  const float* alpha_dn = (const float*)d_in[14];
  const float* beta_up  = (const float*)d_in[15];
  const float* beta_dn  = (const float*)d_in[16];
  const float* mhc_nw   = (const float*)d_in[17];
  const float* phi_pre  = (const float*)d_in[18];
  const float* phi_post = (const float*)d_in[19];
  const float* phi_res  = (const float*)d_in[20];
  const float* b_pre    = (const float*)d_in[21];
  const float* b_post   = (const float*)d_in[22];
  const float* b_res    = (const float*)d_in[23];
  const float* a_pre    = (const float*)d_in[24];
  const float* a_post   = (const float*)d_in[25];
  const float* a_res    = (const float*)d_in[26];
  const float* norm_w   = (const float*)d_in[27];
  const float* W_pre    = (const float*)d_in[28];
  const float* W_o      = (const float*)d_in[29];
  const float* W_pg1    = (const float*)d_in[30];
  const float* W_pg2    = (const float*)d_in[31];
  float* out = (float*)d_out;

  char* p = (char*)d_ws;
  auto alloc = [&](size_t bytes) -> void* {
    void* r = (void*)p; p += (bytes + 255) & ~(size_t)255; return r;
  };
  float* x_hat    = (float*)alloc((size_t)BT * 1024 * 4);
  float* route_in = (float*)alloc((size_t)BT * 256 * 4);
  int*   qsel     = (int*)alloc((size_t)HH * BT * 4);
  int*   kvsel    = (int*)alloc((size_t)HH * BT * 4);
  float* Hpre     = (float*)alloc((size_t)HE * BT * 4 * 4);
  float* Hpost    = (float*)alloc((size_t)HE * BT * 4 * 4);
  float* Hres     = (float*)alloc((size_t)HE * BT * 16 * 4);
  // accumulators (zeroed each launch), contiguous
  float* acc_pre  = (float*)alloc((size_t)BT * 512 * 4);
  float* acc_post = (float*)alloc((size_t)BT * 256 * 4);
  float* acc_res  = (float*)alloc((size_t)BT * 16 * 4);
  float* acc_hp   = (float*)alloc((size_t)BT * 4 * 4);
  float* q_h      = (float*)alloc((size_t)BT * 256 * 4);
  float* k_h      = (float*)alloc((size_t)BT * 256 * 4);
  float* al_h     = (float*)alloc((size_t)BT * 256 * 4);
  float* v_h      = (float*)alloc((size_t)BT * 512 * 4);
  float* be_h     = (float*)alloc((size_t)BT * 8 * 4);
  float* o_ws     = (float*)alloc((size_t)BT * 512 * 4);
  float* result   = (float*)alloc((size_t)BT * 256 * 4);
  bf16_t* gated   = (bf16_t*)alloc((size_t)BT * 512 * 2);
  // packed bf16 weights
  bf16_t* Wpre_b  = (bf16_t*)alloc((size_t)HE * 256 * 512 * 2);
  bf16_t* Wpg1_b  = (bf16_t*)alloc((size_t)HE * 256 * 320 * 2);
  bf16_t* Wpg2_b  = (bf16_t*)alloc((size_t)HE * 320 * 256 * 2);
  bf16_t* Aup_b   = (bf16_t*)alloc((size_t)HE * 256 * 32 * 2);
  bf16_t* Adn_b   = (bf16_t*)alloc((size_t)HE * 32 * 32 * 2);
  bf16_t* Bup_b   = (bf16_t*)alloc((size_t)HE * 256 * 32 * 2);
  bf16_t* Bdn_b   = (bf16_t*)alloc((size_t)HE * 32 * 16 * 2);
  bf16_t* Wqkv_b  = (bf16_t*)alloc((size_t)HH * 256 * 96 * 2);
  bf16_t* Wo_b    = (bf16_t*)alloc((size_t)512 * 256 * 2);
  bf16_t* phiC_b  = (bf16_t*)alloc((size_t)HE * 1024 * 32 * 2);

  auto gsz = [](long n) { return (unsigned)((n + 255) / 256); };

  // zero accumulators (acc_pre..q_h contiguous region)
  {
    long nz = (long)BT * (512 + 256 + 16 + 4) + (long)BT * 256 + 1024;
    hipLaunchKernelGGL(k_zero, dim3(gsz(nz)), dim3(256), 0, stream_, acc_pre, nz);
  }

  hipLaunchKernelGGL(k_prep, dim3(BT), dim3(256), 0, stream_, stream, x_hat, route_in);
  hipLaunchKernelGGL(k_route, dim3(gsz((long)HH * BT)), dim3(256), 0, stream_,
                     route_in, q_router, kv_router, qsel, kvsel);

  // weight packing
  hipLaunchKernelGGL(k_pack, dim3(gsz((long)HE * 256 * 512)), dim3(256), 0, stream_, W_pre, Wpre_b, HE, 256, 512, 256, 512);
  hipLaunchKernelGGL(k_pack, dim3(gsz((long)HE * 256 * 320)), dim3(256), 0, stream_, W_pg1, Wpg1_b, HE, 256, 316, 256, 320);
  hipLaunchKernelGGL(k_pack, dim3(gsz((long)HE * 320 * 256)), dim3(256), 0, stream_, W_pg2, Wpg2_b, HE, 316, 256, 320, 256);
  hipLaunchKernelGGL(k_pack, dim3(gsz((long)HE * 256 * 32)),  dim3(256), 0, stream_, alpha_up, Aup_b, HE, 256, 25, 256, 32);
  hipLaunchKernelGGL(k_pack, dim3(gsz((long)HE * 32 * 32)),   dim3(256), 0, stream_, alpha_dn, Adn_b, HE, 25, 32, 32, 32);
  hipLaunchKernelGGL(k_pack, dim3(gsz((long)HE * 256 * 32)),  dim3(256), 0, stream_, beta_up, Bup_b, HE, 256, 25, 256, 32);
  hipLaunchKernelGGL(k_pack, dim3(gsz((long)HE * 32 * 16)),   dim3(256), 0, stream_, beta_dn, Bdn_b, HE, 25, 1, 32, 16);
  hipLaunchKernelGGL(k_pack, dim3(gsz((long)512 * 256)),      dim3(256), 0, stream_, W_o, Wo_b, 1, 512, 256, 512, 256);
  hipLaunchKernelGGL(k_pack_qkv, dim3(gsz((long)HH * 256 * 96)), dim3(256), 0, stream_, Wq, Wk, Wv, Wqkv_b);
  hipLaunchKernelGGL(k_pack_phi, dim3(gsz((long)HE * 1024 * 32)), dim3(256), 0, stream_,
                     mhc_nw, phi_pre, phi_post, phi_res, phiC_b);

  hipLaunchKernelGGL(k_coef, dim3(BT / 32, HE), dim3(256), 0, stream_,
                     x_hat, phiC_b, a_pre, b_pre, a_post, b_post, a_res, b_res,
                     Hpre, Hpost, Hres);

  hipLaunchKernelGGL(k_expert, dim3(BT / 32, HE), dim3(256), 0, stream_,
                     stream, Hpre, Hres, Hpost, qsel, kvsel, norm_w, pope_d,
                     Wpre_b, Wpg1_b, Wpg2_b, Aup_b, Adn_b, Bup_b, Bdn_b, Wqkv_b,
                     lAq, lBq, lAk, lBk, lAv, lBv,
                     acc_pre, acc_post, acc_res, acc_hp,
                     q_h, k_h, v_h, al_h, be_h);

  hipLaunchKernelGGL(k_scan, dim3(BQ * HH), dim3(64), 0, stream_,
                     q_h, k_h, v_h, al_h, be_h, qsel, o_ws);

  hipLaunchKernelGGL(k_gated, dim3(gsz((long)BT * 512)), dim3(256), 0, stream_,
                     o_ws, acc_pre, gated);
  hipLaunchKernelGGL(k_outgemm, dim3(BT / 32), dim3(256), 0, stream_,
                     gated, Wo_b, acc_post, result);
  hipLaunchKernelGGL(k_mix, dim3(gsz((long)BQ * NMHC * TT * DD)), dim3(256), 0, stream_,
                     stream, acc_res, acc_hp, result, out);
  (void)in_sizes; (void)n_in; (void)out_size; (void)ws_size;
}
